// AttLayer_7103875908022
// MI455X (gfx1250) — matrix operand order repaired and checked by reading, not compiled, not run on hardware
//
#include <hip/hip_runtime.h>

typedef __attribute__((ext_vector_type(16))) __bf16 v16bf;
typedef __attribute__((ext_vector_type(8)))  float  v8f;
typedef __attribute__((ext_vector_type(4)))  float  f4;

#define B_     32
#define S_     2048
#define D_     1024
#define A_     256
#define SCHUNK 128
#define NCHUNK (S_ / SCHUNK)   /* 16 */
#define EPS_   1e-7f

// ---------------------------------------------------------------------------
// Pack W[D,A] (f32 row-major) into WMMA B-fragment-native bf16 layout:
// for tile t (N = t*16..t*16+15), k-tile kt (K = kt*32..kt*32+31), lane l:
//   element i (0..15) holds W[K = kt*32 + (l>>4)*16 + i][N = t*16 + (l&15)]
// stored contiguously: 32 bytes per lane, fully coalesced on load.
// ---------------------------------------------------------------------------
__global__ void wpack_kernel(const float* __restrict__ W,
                             unsigned short* __restrict__ wp) {
    int e    = blockIdx.x * blockDim.x + threadIdx.x;   // 0 .. 262143
    int i    = e & 15;
    int lane = (e >> 4) & 31;
    int kt   = (e >> 9) & 31;
    int tile = e >> 14;
    int N = tile * 16 + (lane & 15);
    int K = kt * 32 + ((lane >> 4) << 4) + i;
    unsigned u = __float_as_uint(W[K * A_ + N]);
    unsigned r = u + 0x7FFFu + ((u >> 16) & 1u);        // RNE f32 -> bf16
    wp[e] = (unsigned short)(r >> 16);
}

// ---------------------------------------------------------------------------
// Fused main kernel: one workgroup = (batch b, 128-row S chunk), 8 waves.
// Phase A: scores = exp(tanh(x W + b) . u) * mask   (bf16 WMMA, f32 acc)
// Phase B: partial numerator Sum_s x[s,:]*score[s] and partial denominator.
// All reductions fixed-order -> bitwise deterministic.
// ---------------------------------------------------------------------------
__global__ __launch_bounds__(256) void attn_main(
        const float* __restrict__ x, const unsigned char* __restrict__ mask,
        const float* __restrict__ bias, const float* __restrict__ u,
        const unsigned short* __restrict__ wp,
        float* __restrict__ pnum, float* __restrict__ pden) {

    __shared__ __attribute__((aligned(16))) float lds_acc[8 * D_]; // 32 KB
    __shared__ float lds_den[8];

    const int b     = blockIdx.y;
    const int chunk = blockIdx.x;
    const int tid   = threadIdx.x;
    const int wave  = tid >> 5;
    const int lane  = tid & 31;
    const int half  = lane >> 4;
    const int lrow  = lane & 15;
    const int rowBase = chunk * SCHUNK + wave * 16;

    const float* xrow = x + ((size_t)b * S_ + rowBase + lrow) * D_;

    float part[8];
#pragma unroll
    for (int r = 0; r < 8; ++r) part[r] = 0.0f;

    // -------- Phase A: GEMM (two passes of 8 N-tiles) --------
    for (int pass = 0; pass < 2; ++pass) {
        v8f c[8] = {};
        for (int kt = 0; kt < 32; ++kt) {
            // A-fragment: row M = lrow, K runs contiguous per half, as in packed W:
            //  i=0..7  -> K = kt*32 + half*16 + i
            //  i=8..15 -> K = kt*32 + half*16 + 8 + (i-8)
            const f4* xp = (const f4*)(xrow + kt * 32 + half * 16);
            const f4* xq = (const f4*)(xrow + kt * 32 + half * 16 + 8);
            f4 a0 = xp[0], a1 = xp[1], a2 = xq[0], a3 = xq[1];
            v16bf af;
            af[0]  = (__bf16)a0.x; af[1]  = (__bf16)a0.y;
            af[2]  = (__bf16)a0.z; af[3]  = (__bf16)a0.w;
            af[4]  = (__bf16)a1.x; af[5]  = (__bf16)a1.y;
            af[6]  = (__bf16)a1.z; af[7]  = (__bf16)a1.w;
            af[8]  = (__bf16)a2.x; af[9]  = (__bf16)a2.y;
            af[10] = (__bf16)a2.z; af[11] = (__bf16)a2.w;
            af[12] = (__bf16)a3.x; af[13] = (__bf16)a3.y;
            af[14] = (__bf16)a3.z; af[15] = (__bf16)a3.w;
#pragma unroll
            for (int t = 0; t < 8; ++t) {
                int tile = pass * 8 + t;
                const f4* wb = (const f4*)(wp + ((((size_t)tile * 32 + kt) * 32 + lane) << 4));
                union { v16bf v; f4 f[2]; } bb;
                bb.f[0] = wb[0];
                bb.f[1] = wb[1];
                c[t] = __builtin_amdgcn_wmma_f32_16x16x32_bf16(
                           false, af, false, bb.v, (short)0, c[t], false, false);
            }
        }
        // fold tiles: part[r] += tanh(c + bias[N]) * u[N], N = tile*16 + lrow
#pragma unroll
        for (int t = 0; t < 8; ++t) {
            int tile = pass * 8 + t;
            float bv = bias[tile * 16 + lrow];
            float uv = u[tile * 16 + lrow];
#pragma unroll
            for (int r = 0; r < 8; ++r)
                part[r] += tanhf(c[t][r] + bv) * uv;
        }
    }

    // -------- reduce over N (16 lanes of each half), exp, mask --------
    float score[8];
#pragma unroll
    for (int r = 0; r < 8; ++r) {
        float v = part[r];
        v += __shfl_xor(v, 1, 32);
        v += __shfl_xor(v, 2, 32);
        v += __shfl_xor(v, 4, 32);
        v += __shfl_xor(v, 8, 32);
        // lanes of half h now hold score-sum for row M = r + h*8
        int srow = rowBase + r + half * 8;
        float m = (float)mask[(size_t)b * S_ + srow];
        score[r] = expf(v) * m;
    }
    float dsum = 0.0f;
#pragma unroll
    for (int r = 0; r < 8; ++r) dsum += score[r];
    float wden = __shfl(dsum, 0, 32) + __shfl(dsum, 16, 32);
    if (lane == 0) lds_den[wave] = wden;

    // -------- Phase B: weighted sum over this wave's 16 rows (L2 re-read) ---
    f4 acc[8] = {};
    const float* xbase = x + ((size_t)b * S_ + rowBase) * D_;
    for (int rr = 0; rr < 16; ++rr) {
        float sr = __shfl(score[rr & 7], (rr >> 3) << 4, 32);
        const f4* xr4 = (const f4*)(xbase + (size_t)rr * D_);
#pragma unroll
        for (int it = 0; it < 8; ++it)
            acc[it] += xr4[it * 32 + lane] * sr;
    }
#pragma unroll
    for (int it = 0; it < 8; ++it)
        *(f4*)&lds_acc[wave * D_ + it * 128 + lane * 4] = acc[it];
    __syncthreads();

    // -------- combine 8 waves (ordered) and emit per-chunk partials --------
    {
        int col = tid * 4;
        f4 s = *(f4*)&lds_acc[col];
#pragma unroll
        for (int w = 1; w < 8; ++w) s += *(f4*)&lds_acc[w * D_ + col];
        *(f4*)(pnum + ((size_t)(b * NCHUNK + chunk)) * D_ + col) = s;
        if (tid == 0) {
            float dd = 0.0f;
            for (int w = 0; w < 8; ++w) dd += lds_den[w];
            pden[b * NCHUNK + chunk] = dd;
        }
    }
}

// ---------------------------------------------------------------------------
// Finalize: out[b,d] = Sum_chunk pnum / (Sum_chunk pden + EPS)
// ---------------------------------------------------------------------------
__global__ void attn_final(const float* __restrict__ pnum,
                           const float* __restrict__ pden,
                           float* __restrict__ out) {
    int b = blockIdx.x, tid = threadIdx.x;       // 256 threads, 4 cols each
    float den = 0.0f;
    for (int c = 0; c < NCHUNK; ++c) den += pden[b * NCHUNK + c];
    float inv = 1.0f / (den + EPS_);
    int col = tid * 4;
    f4 s = {};
    for (int c = 0; c < NCHUNK; ++c)
        s += *(const f4*)&pnum[((size_t)(b * NCHUNK + c)) * D_ + col];
    *(f4*)&out[(size_t)b * D_ + col] = s * inv;
}

extern "C" void kernel_launch(void* const* d_in, const int* in_sizes, int n_in,
                              void* d_out, int out_size, void* d_ws, size_t ws_size,
                              hipStream_t stream) {
    const float*         x    = (const float*)d_in[0];
    const unsigned char* mask = (const unsigned char*)d_in[1];
    const float*         W    = (const float*)d_in[2];
    const float*         bias = (const float*)d_in[3];
    const float*         u    = (const float*)d_in[4];

    // workspace layout: [0, 512KB) packed bf16 W | [512KB, +2MB) partial num |
    //                   [+2KB) partial den
    unsigned short* wp   = (unsigned short*)d_ws;
    float*          pnum = (float*)((char*)d_ws + 512u * 1024u);
    float*          pden = (float*)((char*)d_ws + 512u * 1024u + 2u * 1024u * 1024u);

    wpack_kernel<<<1024, 256, 0, stream>>>(W, wp);
    dim3 grid(NCHUNK, B_);
    attn_main<<<grid, 256, 0, stream>>>(x, mask, bias, u, wp, pnum, pden);
    attn_final<<<B_, 256, 0, stream>>>(pnum, pden, (float*)d_out);
}
